// Attention_89670327206584
// MI455X (gfx1250) — matrix operand order repaired and checked by reading, not compiled, not run on hardware
//
#include <hip/hip_runtime.h>
#include <math.h>

// ---------------------------------------------------------------------------
// Types
// ---------------------------------------------------------------------------
typedef __bf16 bf16;
typedef __attribute__((ext_vector_type(16))) __bf16 bf16x16;
typedef __attribute__((ext_vector_type(8)))  __bf16 bf16x8;
typedef __attribute__((ext_vector_type(8)))  float  f32x8;

// Problem constants
#define N_TOK     4096      // 2 * 2048
#define SEQ       2048
#define DMODEL    1024
#define HEADS     8
#define DHEAD     64
#define ATTN_IN   512       // HEADS*DHEAD
#define FF_IN     4096
#define FUSED_N   8832      // 512 + 64 + 64 + 2*4096

#if defined(__HIP_DEVICE_COMPILE__) && __has_builtin(__builtin_amdgcn_tensor_load_to_lds) && __has_builtin(__builtin_amdgcn_s_wait_tensorcnt)
#define USE_TDM 1
#else
#define USE_TDM 0
#endif

// ---------------------------------------------------------------------------
// WMMA helpers (CDNA5 wave32, V_WMMA_F32_16X16X32_BF16)
// ---------------------------------------------------------------------------
__device__ __forceinline__ f32x8 wmma_bf16(bf16x16 a, bf16x16 b, f32x8 c) {
  return __builtin_amdgcn_wmma_f32_16x16x32_bf16(false, a, false, b,
                                                 (short)0, c, false, false);
}

// A-fragment (16x32 bf16, row-major source, `ld` in elements).
// Lanes 0-15: row M=lane, K = 0..15; lanes 16-31: row M=lane-16,
// K = 16..31  (matches load_b_frag).
__device__ __forceinline__ bf16x16 load_a_frag(const bf16* base, int ld, int lane) {
  int m  = lane & 15;
  int kb = (lane < 16) ? 0 : 16;
  const bf16* p = base + (size_t)m * ld + kb;
  bf16x8 lo = *(const bf16x8*)(p);
  bf16x8 hi = *(const bf16x8*)(p + 8);
  bf16x16 f;
#pragma unroll
  for (int i = 0; i < 8; ++i) { f[i] = lo[i]; f[i + 8] = hi[i]; }
  return f;
}

// B-fragment (32x16): lane's column n = lane&15; lanes 0-15 hold K=0..15,
// lanes 16-31 hold K=16..31. `p` points at this lane's contiguous K-run.
__device__ __forceinline__ bf16x16 load_b_frag(const bf16* p) {
  bf16x8 lo = *(const bf16x8*)(p);
  bf16x8 hi = *(const bf16x8*)(p + 8);
  bf16x16 f;
#pragma unroll
  for (int i = 0; i < 8; ++i) { f[i] = lo[i]; f[i + 8] = hi[i]; }
  return f;
}

// ---------------------------------------------------------------------------
// TDM: 2D tile load Global->LDS (D# per CDNA5 ISA).
// data_size=2B; optional LDS row padding via pad_interval/pad_amount.
// 6-arg builtin: (uint32x4 g0, int32x8 g1, int32x4, int32x4, int32x8, i32 cpol)
// ---------------------------------------------------------------------------
#if USE_TDM
typedef unsigned int v4u  __attribute__((ext_vector_type(4)));
typedef int          v8i_ __attribute__((ext_vector_type(8)));
typedef int          v4i_ __attribute__((ext_vector_type(4)));

__device__ __forceinline__ void tdm_load_2d(
    unsigned lds_addr, const bf16* gptr,
    unsigned tile_k,       // tile_dim0 (elements per row)
    unsigned tile_rows,    // tile_dim1
    unsigned tensor_k,     // tensor_dim0 (elements)
    unsigned tensor_rows,  // tensor_dim1
    unsigned stride_elems, // tensor_dim0_stride
    unsigned pad_elems)    // extra bf16 per LDS row (0 = none)
{
  unsigned long long ga = (unsigned long long)gptr;
  v4u g0;
  g0[0] = 1u;                                            // count=1, user D#
  g0[1] = lds_addr;                                      // lds_addr[31:0]
  g0[2] = (unsigned)ga;                                  // global_addr[31:0]
  g0[3] = (unsigned)((ga >> 32) & 0x1FFFFFFu) | (2u << 30); // ga[56:32], type=2

  unsigned d0 = (1u << 16);                              // data_size = 2B
  if (pad_elems) {
    unsigned row_dwords = tile_k >> 1;                   // bf16 pairs
    unsigned interval_code = (31u - __builtin_clz(row_dwords)) - 1u; // 2^(c+1)
    unsigned pad_dwords = pad_elems >> 1;
    d0 |= (1u << 20) | (interval_code << 22) | ((pad_dwords - 1u) << 25);
  }
  v8i_ g1;
  g1[0] = (int)d0;
  g1[1] = (int)((tensor_k & 0xFFFFu) << 16);             // tensor_dim0[15:0]
  g1[2] = (int)((tensor_k >> 16) | ((tensor_rows & 0xFFFFu) << 16));
  g1[3] = (int)((tensor_rows >> 16) | (tile_k << 16));   // tile_dim0
  g1[4] = (int)(tile_rows & 0xFFFFu);                    // tile_dim1 (dim2=0)
  g1[5] = (int)stride_elems;                             // dim0_stride[31:0]
  g1[6] = 0;                                             // stride hi / dim1_stride
  g1[7] = 0;
  v4i_ z4 = {0, 0, 0, 0};
  v8i_ z8 = {0, 0, 0, 0, 0, 0, 0, 0};
  __builtin_amdgcn_tensor_load_to_lds(g0, g1, z4, z4, z8, 0);
}
#define LDS_OFF(p) ((unsigned)(unsigned long long)(const void*)(p))
#endif

// ---------------------------------------------------------------------------
// Weight convert + transpose: f32 [K][N] -> bf16 [N][K]  (32x32 LDS tiles)
// ---------------------------------------------------------------------------
__global__ __launch_bounds__(256) void cvt_transpose_kernel(
    const float* __restrict__ in, bf16* __restrict__ out, int K, int N) {
  __shared__ float tile[32][33];
  const int k0 = blockIdx.y * 32;
  const int n0 = blockIdx.x * 32;
  const int tx = threadIdx.x & 31;
  const int ty = threadIdx.x >> 5;      // 0..7
#pragma unroll
  for (int i = 0; i < 4; ++i)
    tile[ty + 8 * i][tx] = in[(size_t)(k0 + ty + 8 * i) * N + n0 + tx];
  __syncthreads();
#pragma unroll
  for (int i = 0; i < 4; ++i)
    out[(size_t)(n0 + ty + 8 * i) * K + k0 + tx] = (bf16)tile[tx][ty + 8 * i];
}

// ---------------------------------------------------------------------------
// LayerNorm (no beta) + gamma, f32 -> bf16. One token per block.
// ---------------------------------------------------------------------------
__global__ __launch_bounds__(256) void ln_kernel(
    const float* __restrict__ x, const float* __restrict__ gamma,
    bf16* __restrict__ xn) {
  const int t = blockIdx.x;
  const float* xr = x + (size_t)t * DMODEL;
  __shared__ float red0[8], red1[8];

  float s = 0.f, ss = 0.f;
  for (int j = threadIdx.x; j < DMODEL; j += 256) {
    float v = xr[j]; s += v; ss += v * v;
  }
  int lane = threadIdx.x & 31, wid = threadIdx.x >> 5;
#pragma unroll
  for (int m = 16; m >= 1; m >>= 1) {
    s  += __shfl_xor(s,  m, 32);
    ss += __shfl_xor(ss, m, 32);
  }
  if (lane == 0) { red0[wid] = s; red1[wid] = ss; }
  __syncthreads();
  float tot = 0.f, tot2 = 0.f;
#pragma unroll
  for (int w = 0; w < 8; ++w) { tot += red0[w]; tot2 += red1[w]; }
  float mu  = tot * (1.0f / DMODEL);
  float var = tot2 * (1.0f / DMODEL) - mu * mu;
  float rs  = rsqrtf(var + 1e-5f);
  bf16* xo = xn + (size_t)t * DMODEL;
  for (int j = threadIdx.x; j < DMODEL; j += 256)
    xo[j] = (bf16)((xr[j] - mu) * rs * gamma[j]);
}

// ---------------------------------------------------------------------------
// Tiled bf16 WMMA GEMM: C[M,N](f32) = A[M,K] @ B[K,N], B given TRANSPOSED
// (Bt[N][K]). Block tile 256x128, K-stage 64; 8 waves (4x2), 64x64 per wave
// = 4x4 frags, 32 WMMAs per stage. TDM double-buffered pipeline when
// available: stage k+1's DMA overlaps stage k's WMMA work.
// ---------------------------------------------------------------------------
#define GBM 256
#define GBN 128
#define GBK 64
#define A_STR 72   // 64 + 8 pad (bank-stagger); 144B rows
#define B_STR 72

__global__ __launch_bounds__(256) void gemm_bf16_kernel(
    const bf16* __restrict__ A, const bf16* __restrict__ Bt,
    float* __restrict__ C, int M, int N, int K, int accumulate) {
  __shared__ bf16 As[2][GBM * A_STR];   // 2 x 36 KB
  __shared__ bf16 Bs[2][GBN * B_STR];   // 2 x 18 KB

  const int tid  = threadIdx.x;
  const int lane = tid & 31;
  const int wid  = tid >> 5;
  const int wm   = wid >> 1;     // 0..3
  const int wn   = wid & 1;      // 0..1
  const int row0 = blockIdx.y * GBM;
  const int col0 = blockIdx.x * GBN;
  const int wrow = wm * 64;
  const int wcol = wn * 64;
  const int ln   = lane & 15;
  const int kb2  = (lane < 16) ? 0 : 16;

#if USE_TDM
  // LDS is written by TDM (which only sees integer addresses) -- force the
  // compiler to treat the arrays as escaped/modified.
  asm volatile("" : : "v"(&As[0][0]), "v"(&Bs[0][0]) : "memory");
#endif

  f32x8 acc[4][4];
#pragma unroll
  for (int i = 0; i < 4; ++i)
#pragma unroll
    for (int j = 0; j < 4; ++j) acc[i][j] = (f32x8){0,0,0,0,0,0,0,0};

#if USE_TDM
  const int nstages = K / GBK;
  if (wid == 0) {   // prologue: stage 0 into buffer 0
    tdm_load_2d(LDS_OFF(As[0]), A  + (size_t)row0 * K, GBK, GBM,
                (unsigned)K, (unsigned)M, (unsigned)K, A_STR - GBK);
    tdm_load_2d(LDS_OFF(Bs[0]), Bt + (size_t)col0 * K, GBK, GBN,
                (unsigned)K, (unsigned)N, (unsigned)K, B_STR - GBK);
  }
  for (int ks = 0; ks < nstages; ++ks) {
    const int cur = ks & 1;
    if (wid == 0 && (ks + 1) < nstages) {
      // buf[cur^1] was last read at stage ks-1; end-of-stage barrier passed.
      int k1 = (ks + 1) * GBK;
      tdm_load_2d(LDS_OFF(As[cur ^ 1]), A  + (size_t)row0 * K + k1, GBK, GBM,
                  (unsigned)K, (unsigned)M, (unsigned)K, A_STR - GBK);
      tdm_load_2d(LDS_OFF(Bs[cur ^ 1]), Bt + (size_t)col0 * K + k1, GBK, GBN,
                  (unsigned)K, (unsigned)N, (unsigned)K, B_STR - GBK);
    }
    // Tensor ops complete in order: <=2 outstanding == current stage landed.
    if ((ks + 1) < nstages) __builtin_amdgcn_s_wait_tensorcnt(2);
    else                    __builtin_amdgcn_s_wait_tensorcnt(0);
    __syncthreads();

    const bf16* as = As[cur];
    const bf16* bs = Bs[cur];
#pragma unroll
    for (int s = 0; s < 2; ++s) {
      bf16x16 af[4], bfr[4];
#pragma unroll
      for (int mf = 0; mf < 4; ++mf)
        af[mf] = load_a_frag(&as[(wrow + mf * 16) * A_STR + 32 * s], A_STR, lane);
#pragma unroll
      for (int nf = 0; nf < 4; ++nf)
        bfr[nf] = load_b_frag(&bs[(wcol + nf * 16 + ln) * B_STR + 32 * s + kb2]);
#pragma unroll
      for (int mf = 0; mf < 4; ++mf)
#pragma unroll
        for (int nf = 0; nf < 4; ++nf)
          acc[mf][nf] = wmma_bf16(af[mf], bfr[nf], acc[mf][nf]);
    }
    __syncthreads();
  }
#else
  for (int k0 = 0; k0 < K; k0 += GBK) {
    {   // A tile 256x64: one row per thread, 8 x 16B contiguous
      const bf16* src = A + (size_t)(row0 + tid) * K + k0;
#pragma unroll
      for (int i = 0; i < 8; ++i)
        *(bf16x8*)(&As[0][tid * A_STR + 8 * i]) = *(const bf16x8*)(src + 8 * i);
    }
    {   // Bt tile 128x64: half row per thread, 4 x 16B contiguous
      int r  = tid >> 1;
      int c0 = (tid & 1) * 32;
      const bf16* src = Bt + (size_t)(col0 + r) * K + k0 + c0;
#pragma unroll
      for (int i = 0; i < 4; ++i)
        *(bf16x8*)(&Bs[0][r * B_STR + c0 + 8 * i]) = *(const bf16x8*)(src + 8 * i);
    }
    __syncthreads();
#pragma unroll
    for (int s = 0; s < 2; ++s) {
      bf16x16 af[4], bfr[4];
#pragma unroll
      for (int mf = 0; mf < 4; ++mf)
        af[mf] = load_a_frag(&As[0][(wrow + mf * 16) * A_STR + 32 * s], A_STR, lane);
#pragma unroll
      for (int nf = 0; nf < 4; ++nf)
        bfr[nf] = load_b_frag(&Bs[0][(wcol + nf * 16 + ln) * B_STR + 32 * s + kb2]);
#pragma unroll
      for (int mf = 0; mf < 4; ++mf)
#pragma unroll
        for (int nf = 0; nf < 4; ++nf)
          acc[mf][nf] = wmma_bf16(af[mf], bfr[nf], acc[mf][nf]);
    }
    __syncthreads();
  }
#endif

  // Epilogue: branch hoisted out of the store loops.
  const int rbase = (lane < 16) ? 0 : 8;
  if (accumulate) {
#pragma unroll
    for (int mf = 0; mf < 4; ++mf)
#pragma unroll
      for (int nf = 0; nf < 4; ++nf) {
        int col = col0 + wcol + nf * 16 + ln;
#pragma unroll
        for (int r = 0; r < 8; ++r) {
          size_t idx = (size_t)(row0 + wrow + mf * 16 + rbase + r) * N + col;
          C[idx] += acc[mf][nf][r];
        }
      }
  } else {
#pragma unroll
    for (int mf = 0; mf < 4; ++mf)
#pragma unroll
      for (int nf = 0; nf < 4; ++nf) {
        int col = col0 + wcol + nf * 16 + ln;
#pragma unroll
        for (int r = 0; r < 8; ++r) {
          size_t idx = (size_t)(row0 + wrow + mf * 16 + rbase + r) * N + col;
          C[idx] = acc[mf][nf][r];
        }
      }
  }
}

// ---------------------------------------------------------------------------
// Split proj -> q (scaled, [b][h][n][64]), k ([b][n][64]), vT ([b][64][n]),
// h = silu(gate)*ff_x ([t][4096]); all bf16.
// ---------------------------------------------------------------------------
#define SPLIT_COLS (640 + 4096)

__global__ __launch_bounds__(256) void split_kernel(
    const float* __restrict__ proj, bf16* __restrict__ q, bf16* __restrict__ k,
    bf16* __restrict__ vT, bf16* __restrict__ h) {
  size_t idx = (size_t)blockIdx.x * 256 + threadIdx.x;
  if (idx >= (size_t)N_TOK * SPLIT_COLS) return;
  int t = (int)(idx / SPLIT_COLS);
  int c = (int)(idx % SPLIT_COLS);
  int b = t >> 11, n = t & 2047;
  const float* pr = proj + (size_t)t * FUSED_N;

  if (c < ATTN_IN) {                       // Q, fold 1/sqrt(DHEAD)
    int hd = c >> 6, d = c & 63;
    q[((((size_t)b * HEADS + hd) * SEQ) + n) * DHEAD + d] = (bf16)(pr[c] * 0.125f);
  } else if (c < ATTN_IN + DHEAD) {        // K row-major
    k[(((size_t)b * SEQ) + n) * DHEAD + (c - ATTN_IN)] = (bf16)pr[c];
  } else if (c < ATTN_IN + 2 * DHEAD) {    // V transposed
    int d = c - (ATTN_IN + DHEAD);
    vT[(((size_t)b * DHEAD) + d) * SEQ + n] = (bf16)pr[c];
  } else {                                 // h = silu(gate) * ff_x
    int i = c - (ATTN_IN + 2 * DHEAD);
    float ffx = pr[ATTN_IN + 2 * DHEAD + i];
    float g   = pr[ATTN_IN + 2 * DHEAD + FF_IN + i];
    float sg  = g / (1.0f + __expf(-g));
    h[(size_t)t * FF_IN + i] = (bf16)(ffx * sg);
  }
}

// ---------------------------------------------------------------------------
// Flash attention (MQA). Grid (16 qblocks, 8 heads, 2 batches), 8 waves;
// each wave: 16 query rows. K/V tiles are staged ONCE per workgroup into LDS
// (TDM when available) -- all 8 waves share them, cutting 8x-redundant
// global traffic. WMMA for S=QK^T and O+=PV; online softmax in f32.
// ---------------------------------------------------------------------------
#define KT 128
#define P_STR 144
#define K_STR 72    // 64 + 8 pad
#define V_STR 136   // 128 + 8 pad

__global__ __launch_bounds__(256) void attn_kernel(
    const bf16* __restrict__ q, const bf16* __restrict__ k,
    const bf16* __restrict__ vT, bf16* __restrict__ out) {
  __shared__ bf16 Kt[KT * K_STR];        // 18 KB, rows = key position j
  __shared__ bf16 Vt[DHEAD * V_STR];     // 17 KB, rows = head dim d
  __shared__ bf16 Pl[8][16 * P_STR];     // 36 KB, per-wave P staging

  const int tid  = threadIdx.x;
  const int lane = tid & 31;
  const int w    = tid >> 5;
  const int qb   = blockIdx.x;
  const int hd   = blockIdx.y;
  const int b    = blockIdx.z;
  const int qr0  = qb * 128 + w * 16;

  const bf16* qbase = q  + ((((size_t)b * HEADS + hd) * SEQ) + qr0) * DHEAD;
  const bf16* kbase = k  + ((size_t)b * SEQ) * DHEAD;
  const bf16* vbase = vT + ((size_t)b * DHEAD) * SEQ;
  bf16* pw = &Pl[w][0];

#if USE_TDM
  asm volatile("" : : "v"(&Kt[0]), "v"(&Vt[0]) : "memory");
#endif

  bf16x16 aq[2];
#pragma unroll
  for (int s = 0; s < 2; ++s) aq[s] = load_a_frag(qbase + 32 * s, DHEAD, lane);

  f32x8 o[4];
#pragma unroll
  for (int vn = 0; vn < 4; ++vn) o[vn] = (f32x8){0,0,0,0,0,0,0,0};
  float mrow[8], lrow[8];
#pragma unroll
  for (int r = 0; r < 8; ++r) { mrow[r] = -3.0e38f; lrow[r] = 0.f; }

  const int ln  = lane & 15;
  const int kb2 = (lane < 16) ? 0 : 16;

  for (int jt = 0; jt < SEQ; jt += KT) {
    // ---- stage K tile (128 x 64) and V^T tile (64 x 128) into LDS ----
#if USE_TDM
    if (w == 0) {
      tdm_load_2d(LDS_OFF(Kt), kbase + (size_t)jt * DHEAD, DHEAD, KT,
                  DHEAD, SEQ, DHEAD, K_STR - DHEAD);
      tdm_load_2d(LDS_OFF(Vt), vbase + jt, KT, DHEAD,
                  SEQ, DHEAD, SEQ, V_STR - KT);
    }
    __builtin_amdgcn_s_wait_tensorcnt(0);
    __syncthreads();
#else
    {   // K: 128 rows x 64, half row per thread
      int r  = tid >> 1;
      int c0 = (tid & 1) * 32;
      const bf16* src = kbase + (size_t)(jt + r) * DHEAD + c0;
#pragma unroll
      for (int i = 0; i < 4; ++i)
        *(bf16x8*)(&Kt[r * K_STR + c0 + 8 * i]) = *(const bf16x8*)(src + 8 * i);
    }
    {   // V^T: 64 rows x 128, quarter row per thread
      int r  = tid >> 2;
      int c0 = (tid & 3) * 32;
      const bf16* src = vbase + (size_t)r * SEQ + jt + c0;
#pragma unroll
      for (int i = 0; i < 4; ++i)
        *(bf16x8*)(&Vt[r * V_STR + c0 + 8 * i]) = *(const bf16x8*)(src + 8 * i);
    }
    __syncthreads();
#endif

    // ---- S = Q K^T over this 16x128 tile ----
    f32x8 sfr[8];
#pragma unroll
    for (int nf = 0; nf < 8; ++nf) {
      f32x8 a = (f32x8){0,0,0,0,0,0,0,0};
      int j = nf * 16 + ln;     // local key index
#pragma unroll
      for (int s = 0; s < 2; ++s) {
        bf16x16 bk = load_b_frag(&Kt[j * K_STR + 32 * s + kb2]);
        a = wmma_bf16(aq[s], bk, a);
      }
      sfr[nf] = a;
    }
    // ---- online softmax: row stats via 16-lane half-wave shuffles ----
    float scale[8], mnew[8];
#pragma unroll
    for (int r = 0; r < 8; ++r) {
      float mx = sfr[0][r];
#pragma unroll
      for (int nf = 1; nf < 8; ++nf) mx = fmaxf(mx, sfr[nf][r]);
#pragma unroll
      for (int m = 8; m >= 1; m >>= 1) mx = fmaxf(mx, __shfl_xor(mx, m, 16));
      mnew[r]  = fmaxf(mrow[r], mx);
      scale[r] = __expf(mrow[r] - mnew[r]);
      mrow[r]  = mnew[r];
    }
    // ---- P = exp(S - m): C-layout -> A-layout via per-wave LDS tile ----
#pragma unroll
    for (int r = 0; r < 8; ++r) {
      float rs = 0.f;
      int row = ((lane < 16) ? r : r + 8);
#pragma unroll
      for (int nf = 0; nf < 8; ++nf) {
        float p = __expf(sfr[nf][r] - mnew[r]);
        rs += p;
        pw[row * P_STR + nf * 16 + ln] = (bf16)p;
      }
#pragma unroll
      for (int m = 8; m >= 1; m >>= 1) rs += __shfl_xor(rs, m, 16);
      lrow[r] = lrow[r] * scale[r] + rs;
    }
    // ---- rescale O, then O += P @ V ----
#pragma unroll
    for (int vn = 0; vn < 4; ++vn)
#pragma unroll
      for (int r = 0; r < 8; ++r) o[vn][r] *= scale[r];
#pragma unroll
    for (int s2 = 0; s2 < 4; ++s2) {
      bf16x16 ap = load_a_frag(pw + 32 * s2, P_STR, lane);
#pragma unroll
      for (int vn = 0; vn < 4; ++vn) {
        int d = vn * 16 + ln;
        bf16x16 bv = load_b_frag(&Vt[d * V_STR + 32 * s2 + kb2]);
        o[vn] = wmma_bf16(ap, bv, o[vn]);
      }
    }
    __syncthreads();   // all waves done with Kt/Vt before next stage
  }

  const int rbase = (lane < 16) ? 0 : 8;
#pragma unroll
  for (int vn = 0; vn < 4; ++vn)
#pragma unroll
    for (int r = 0; r < 8; ++r) {
      float v = o[vn][r] / lrow[r];
      int row = qr0 + rbase + r;
      out[((size_t)b * SEQ + row) * ATTN_IN + hd * DHEAD + vn * 16 + ln] = (bf16)v;
    }
}

// ---------------------------------------------------------------------------
// Launch
// ---------------------------------------------------------------------------
extern "C" void kernel_launch(void* const* d_in, const int* in_sizes, int n_in,
                              void* d_out, int out_size, void* d_ws, size_t ws_size,
                              hipStream_t stream) {
  const float* x     = (const float*)d_in[0];
  const float* gamma = (const float*)d_in[1];
  const float* wf    = (const float*)d_in[2];  // 1024 x 8832
  const float* wao   = (const float*)d_in[3];  // 512  x 1024
  const float* wfo   = (const float*)d_in[4];  // 4096 x 1024
  float* out = (float*)d_out;

  char* ws = (char*)d_ws;
  size_t off = 0;
  auto take = [&](size_t bytes) { char* p = ws + off; off += (bytes + 255) & ~(size_t)255; return p; };

  bf16* xn_bf  = (bf16*)take((size_t)N_TOK * DMODEL * 2);
  bf16* wfT    = (bf16*)take((size_t)FUSED_N * DMODEL * 2);  // [8832][1024]
  bf16* waoT   = (bf16*)take((size_t)DMODEL * ATTN_IN * 2);  // [1024][512]
  bf16* wfoT   = (bf16*)take((size_t)DMODEL * FF_IN * 2);    // [1024][4096]
  float* proj  = (float*)take((size_t)N_TOK * FUSED_N * 4);
  bf16* q_bf   = (bf16*)take((size_t)2 * HEADS * SEQ * DHEAD * 2);
  bf16* k_bf   = (bf16*)take((size_t)2 * SEQ * DHEAD * 2);
  bf16* vT_bf  = (bf16*)take((size_t)2 * DHEAD * SEQ * 2);
  bf16* h_bf   = (bf16*)take((size_t)N_TOK * FF_IN * 2);
  bf16* ao_bf  = (bf16*)take((size_t)N_TOK * ATTN_IN * 2);

  // 1) weights -> bf16, transposed to [N][K] for contiguous B-fragments
  cvt_transpose_kernel<<<dim3(FUSED_N / 32, DMODEL / 32), 256, 0, stream>>>(
      wf, wfT, DMODEL, FUSED_N);
  cvt_transpose_kernel<<<dim3(DMODEL / 32, ATTN_IN / 32), 256, 0, stream>>>(
      wao, waoT, ATTN_IN, DMODEL);
  cvt_transpose_kernel<<<dim3(DMODEL / 32, FF_IN / 32), 256, 0, stream>>>(
      wfo, wfoT, FF_IN, DMODEL);

  // 2) LayerNorm -> bf16 activations
  ln_kernel<<<N_TOK, 256, 0, stream>>>(x, gamma, xn_bf);

  // 3) proj = xn @ w_fused   (4096 x 1024 x 8832)
  gemm_bf16_kernel<<<dim3(FUSED_N / GBN, N_TOK / GBM), 256, 0, stream>>>(
      xn_bf, wfT, proj, N_TOK, FUSED_N, DMODEL, 0);

  // 4) split into q/k/vT/h
  {
    size_t tot = (size_t)N_TOK * SPLIT_COLS;
    int blocks = (int)((tot + 255) / 256);
    split_kernel<<<blocks, 256, 0, stream>>>(proj, q_bf, k_bf, vT_bf, h_bf);
  }

  // 5) attention
  attn_kernel<<<dim3(SEQ / 128, HEADS, 2), 256, 0, stream>>>(q_bf, k_bf, vT_bf, ao_bf);

  // 6) out = attn_flat @ w_attn_out    (4096 x 512 x 1024)
  gemm_bf16_kernel<<<dim3(DMODEL / GBN, N_TOK / GBM), 256, 0, stream>>>(
      ao_bf, waoT, out, N_TOK, DMODEL, ATTN_IN, 0);

  // 7) out += h @ w_ff_out             (4096 x 4096 x 1024)
  gemm_bf16_kernel<<<dim3(DMODEL / GBN, N_TOK / GBM), 256, 0, stream>>>(
      h_bf, wfoT, out, N_TOK, DMODEL, FF_IN, 1);
}
